// WindowAttention_893353198365
// MI455X (gfx1250) — matrix operand order repaired and checked by reading, not compiled, not run on hardware
//
#include <hip/hip_runtime.h>
#include <hip/hip_bf16.h>

// ---------------------------------------------------------------------------
// Sizes
// ---------------------------------------------------------------------------
#define BB    128
#define HH_   14
#define NN    196       // 14*14 tokens
#define CC    256
#define NHEAD 8
#define HD    32
#define AG    49
#define RR    16
#define SCALE 0.17677669529663689f   // 1/sqrt(32)
#define BASE_STRIDE 50176            // C*N = N*C per batch (flat)

typedef __attribute__((ext_vector_type(16))) _Float16 v16h;
typedef __attribute__((ext_vector_type(8)))  float    v8f;
typedef __attribute__((ext_vector_type(8)))  _Float16 h8;

// ---------------------------------------------------------------------------
// WMMA helpers (CDNA5 wave32, V_WMMA_F32_16X16X32_F16)
// A fragment: 16x32 (MxK) f16.  Lane L: M = m0+(L&15); K block = k0 + (L>=16?8:0),
//   elements e=0..7 -> K=kb..kb+7 ; e=8..15 -> K=kb+16..kb+23   (ISA 7.12.2)
// B fragment: 32x16 (KxN) f16, stored K-major [K][N].  Lane L: N=n0+(L&15); K block
//   kb = k0 + (L>=16?8:0), elements e=0..7 -> K=kb..kb+7 ; e=8..15 -> K=kb+16..kb+23
// C/D: 8 VGPRs f32. Lane L: N=n0+(L&15); VGPR r: M=m0+r+(L>=16?8:0)
// ---------------------------------------------------------------------------
__device__ __forceinline__ v8f wmma_f32(v16h a, v16h b, v8f c) {
  return __builtin_amdgcn_wmma_f32_16x16x32_f16(false, a, false, b, (short)0, c,
                                                false, false);
}

__device__ __forceinline__ v16h ldA(const _Float16* S, int ld, int m0, int k0) {
  int lane = threadIdx.x & 31;
  int M = m0 + (lane & 15);
  int kb = k0 + ((lane >> 4) << 3);
  h8 x0 = *(const h8*)(S + M * ld + kb);
  h8 x1 = *(const h8*)(S + M * ld + kb + 16);
  v16h a;
#pragma unroll
  for (int e = 0; e < 8; ++e) { a[e] = x0[e]; a[e + 8] = x1[e]; }
  return a;
}

// A fragment where each lane supplies its own row index (for implicit-GEMM conv)
__device__ __forceinline__ v16h ldA_row(const _Float16* S, int ld, int row, int k0) {
  int lane = threadIdx.x & 31;
  int kb = k0 + ((lane >> 4) << 3);
  h8 x0 = *(const h8*)(S + row * ld + kb);
  h8 x1 = *(const h8*)(S + row * ld + kb + 16);
  v16h a;
#pragma unroll
  for (int e = 0; e < 8; ++e) { a[e] = x0[e]; a[e + 8] = x1[e]; }
  return a;
}

__device__ __forceinline__ v16h ldB(const _Float16* T, int ld, int k0, int n0) {
  int lane = threadIdx.x & 31;
  const _Float16* p = T + (k0 + ((lane >> 4) << 3)) * ld + n0 + (lane & 15);
  v16h b;
#pragma unroll
  for (int e = 0; e < 8; ++e) { b[e] = p[e * ld]; b[e + 8] = p[(e + 16) * ld]; }
  return b;
}

__device__ __forceinline__ float sigmoidf(float v) {
  return 1.0f / (1.0f + __expf(-v));
}

// bilinear 7 -> 14 grid helper (matches reference grid())
__device__ __forceinline__ void bgrid(int o, int& i0, int& i1, float& l) {
  float pos = fmaxf((o + 0.5f) * 0.5f - 0.5f, 0.0f);
  int f = (int)floorf(pos);
  i0 = f < 6 ? f : 6;
  i1 = (i0 + 1) < 6 ? (i0 + 1) : 6;
  l = pos - (float)i0;
}

// ---------------------------------------------------------------------------
// K1: LKA depthwise 5x5 conv (groups=C, 3 outputs per input channel)
//     x (B,N,C) -> k_raw / v_raw / q_raw  (B,C,14,14) flat
// ---------------------------------------------------------------------------
__global__ void k_lka(const float* __restrict__ x, const float* __restrict__ lw,
                      const float* __restrict__ lb, float* __restrict__ qraw,
                      float* __restrict__ kraw, float* __restrict__ vraw) {
  __shared__ float plane[NN];
  __shared__ float wsm[25];
  int bid = blockIdx.x;
  int b = bid / (3 * CC), o = bid % (3 * CC);
  int ic = o / 3;                         // feature_group_count=C, 3 out per group
  int t = threadIdx.x;
  if (t < NN) plane[t] = x[((size_t)b * NN + t) * CC + ic];
  if (t < 25) wsm[t] = lw[o * 25 + t];
  __syncthreads();
  if (t < NN) {
    int y = t / 14, xx = t % 14;
    float acc = lb[o];
#pragma unroll
    for (int ky = 0; ky < 5; ++ky) {
      int iy = y + ky - 2;
      if (iy < 0 || iy >= 14) continue;
#pragma unroll
      for (int kx = 0; kx < 5; ++kx) {
        int ix = xx + kx - 2;
        if (ix < 0 || ix >= 14) continue;
        acc += plane[iy * 14 + ix] * wsm[ky * 5 + kx];
      }
    }
    float* dst = (o < CC) ? kraw : (o < 2 * CC ? vraw : qraw);
    int oc = o & 255;
    dst[((size_t)b * CC + oc) * NN + t] = acc;
  }
}

// ---------------------------------------------------------------------------
// K2: agent token pooling: at = mean-pool(q_raw) 2x2, atk = max-pool(k_raw)
//     outputs (B, 49, C)
// ---------------------------------------------------------------------------
__global__ void k_pool(const float* __restrict__ qraw, const float* __restrict__ kraw,
                       float* __restrict__ at, float* __restrict__ atk) {
  int tid = blockIdx.x * 256 + threadIdx.x;
  if (tid >= BB * AG * CC) return;
  int cc = tid & 255;
  int a = (tid >> 8) % AG;
  int b = tid / (AG * CC);
  int p1 = a / 7, p2 = a % 7;
  const float* qp = qraw + ((size_t)b * CC + cc) * NN;
  const float* kp = kraw + ((size_t)b * CC + cc) * NN;
  float s = 0.f, m = -1e30f;
#pragma unroll
  for (int g1 = 0; g1 < 2; ++g1)
#pragma unroll
    for (int g2 = 0; g2 < 2; ++g2) {
      int sp = (p1 * 2 + g1) * 14 + (p2 * 2 + g2);
      s += qp[sp];
      m = fmaxf(m, kp[sp]);
    }
  at[((size_t)b * AG + a) * CC + cc] = s * 0.25f;
  atk[((size_t)b * AG + a) * CC + cc] = m;
}

// ---------------------------------------------------------------------------
// K3: conv5 dense 5x5 (256->256) as implicit GEMM with WMMA.
//     workgroup = (b, co-tile of 32). M=196 pos (13 tiles), N=32 co (2 tiles),
//     K = 25 taps x 8 chunks of 32 in-ch.
// ---------------------------------------------------------------------------
__global__ void __launch_bounds__(256) k_conv5(const float* __restrict__ vraw,
                                               const float* __restrict__ w5,
                                               const float* __restrict__ b5,
                                               float* __restrict__ v5) {
  extern __shared__ char smem[];
  _Float16* Xs = (_Float16*)smem;            // [324 padded pos][256 ci]
  _Float16* Wt = Xs + 324 * 256;             // [256 ci][32 co]
  int b = blockIdx.x >> 3;
  int co0 = (blockIdx.x & 7) * 32;
  int tid = threadIdx.x, lane = tid & 31, wave = tid >> 5;
  int hi8 = ((lane >> 4) << 3);
  for (int idx = tid; idx < 324 * 256; idx += 256) {
    int row = idx >> 8, ci = idx & 255;
    int py = row / 18 - 2, px = row % 18 - 2;
    float v = 0.f;
    if (py >= 0 && py < 14 && px >= 0 && px < 14)
      v = vraw[((size_t)b * CC + ci) * NN + py * 14 + px];
    Xs[row * 256 + ci] = (_Float16)v;
  }
  v8f acc[4] = {};
  for (int tap = 0; tap < 25; ++tap) {
    __syncthreads();
    for (int idx = tid; idx < 256 * 32; idx += 256) {
      int ci = idx >> 5, col = idx & 31;
      Wt[ci * 32 + col] = (_Float16)w5[((size_t)(co0 + col) * CC + ci) * 25 + tap];
    }
    __syncthreads();
    int ky = tap / 5, kx = tap % 5;
    for (int j = 0; j < 4; ++j) {
      int t = wave + j * 8;
      if (t >= 26) break;                     // wave-uniform
      int mt = t % 13, nt = t / 13;
      int p = mt * 16 + (lane & 15);
      if (p > 195) p = 195;                   // padded rows never stored
      int prow = (p / 14 + ky) * 18 + (p % 14 + kx);
#pragma unroll
      for (int kc = 0; kc < 8; ++kc) {
        v16h a = ldA_row(Xs, 256, prow, kc * 32);
        v16h bf = ldB(Wt, 32, kc * 32, nt * 16);
        acc[j] = wmma_f32(a, bf, acc[j]);
      }
    }
  }
  for (int j = 0; j < 4; ++j) {
    int t = wave + j * 8;
    if (t >= 26) break;
    int mt = t % 13, nt = t / 13;
    int co = co0 + nt * 16 + (lane & 15);
    float bias = b5[co];
#pragma unroll
    for (int r = 0; r < 8; ++r) {
      int m = mt * 16 + r + hi8;
      if (m < NN) v5[((size_t)b * CC + co) * NN + m] = acc[j][r] + bias;
    }
  }
}

// ---------------------------------------------------------------------------
// K4: CBAM channel attention -> sigc[b][c]
// ---------------------------------------------------------------------------
__global__ void k_cbam_ch(const float* __restrict__ v5, const float* caa1w,
                          const float* caa1b, const float* caa2w, const float* caa2b,
                          const float* cam1w, const float* cam1b, const float* cam2w,
                          const float* cam2b, float* __restrict__ sigc) {
  __shared__ float avg[CC], mx[CC], t1[RR], t2[RR];
  int b = blockIdx.x, c = threadIdx.x;
  const float* p = v5 + ((size_t)b * CC + c) * NN;
  float s = 0.f, m = -1e30f;
  for (int i = 0; i < NN; ++i) { float v = p[i]; s += v; m = fmaxf(m, v); }
  avg[c] = s * (1.f / 196.f);
  mx[c] = m;
  __syncthreads();
  if (c < RR) {
    float a1 = caa1b[c], a2 = cam1b[c];
    for (int j = 0; j < CC; ++j) {
      a1 += caa1w[(c * CC + j) * 9 + 4] * avg[j];   // 3x3 center tap
      a2 += cam1w[c * CC + j] * mx[j];              // 1x1
    }
    t1[c] = fmaxf(a1, 0.f);
    t2[c] = fmaxf(a2, 0.f);
  }
  __syncthreads();
  float y1 = caa2b[c], y2 = cam2b[c];
  for (int r = 0; r < RR; ++r) {
    y1 += caa2w[(c * RR + r) * 9 + 4] * t1[r];
    y2 += cam2w[c * RR + r] * t2[r];
  }
  sigc[b * CC + c] = sigmoidf(sigmoidf(y1) + sigmoidf(y2));
}

// ---------------------------------------------------------------------------
// K5: CBAM spatial attention; overwrites v5 with vc = v5*sigc*mask
// ---------------------------------------------------------------------------
__global__ void k_cbam_sp(float* __restrict__ v5, const float* __restrict__ sigc,
                          const float* __restrict__ saw, const float* __restrict__ sab) {
  __shared__ float mn[NN], mxs[NN], mask[NN], sc[CC];
  int b = blockIdx.x, t = threadIdx.x;
  sc[t] = sigc[b * CC + t];
  __syncthreads();
  if (t < NN) {
    float s = 0.f, m = -1e30f;
    for (int c = 0; c < CC; ++c) {
      float v = v5[((size_t)b * CC + c) * NN + t] * sc[c];
      s += v;
      m = fmaxf(m, v);
    }
    mn[t] = s * (1.f / 256.f);
    mxs[t] = m;
  }
  __syncthreads();
  if (t < NN) {
    int y = t / 14, x = t % 14;
    float acc = sab[0];
    for (int ky = 0; ky < 9; ++ky) {
      int iy = y + ky - 4;
      if (iy < 0 || iy >= 14) continue;
      for (int kx = 0; kx < 9; ++kx) {
        int ix = x + kx - 4;
        if (ix < 0 || ix >= 14) continue;
        acc += mn[iy * 14 + ix] * saw[ky * 9 + kx] + mxs[iy * 14 + ix] * saw[81 + ky * 9 + kx];
      }
    }
    mask[t] = sigmoidf(acc);
  }
  __syncthreads();
  for (int idx = t; idx < CC * NN; idx += 256) {
    int c = idx / NN, s = idx % NN;
    v5[((size_t)b * CC + c) * NN + s] *= sc[c] * mask[s];
  }
}

// ---------------------------------------------------------------------------
// K6: window attention: x1 = softmax(Q K^T + rpb) V.  One WG per (b,head).
//     Whole 196x196 score matrix lives in LDS (f32), P converted to f16.
// ---------------------------------------------------------------------------
__global__ void __launch_bounds__(256) k_attn(const float* __restrict__ qraw,
                                              const float* __restrict__ kraw,
                                              const float* __restrict__ vraw,
                                              const float* __restrict__ rpbt,
                                              float* __restrict__ x1) {
  extern __shared__ char smem[];
  _Float16* Qs = (_Float16*)smem;        // 208x32
  _Float16* Kt = Qs + 208 * 32;          // 32x208 (K transposed: [hd][token])
  _Float16* Vs = Kt + 32 * 208;          // 224x32
  _Float16* Pf = Vs + 224 * 32;          // 208x224 probs f16
  float* Sf = (float*)(Pf + 208 * 224);  // 208x208 logits f32
  float* rc = Sf + 208 * 208;            // 729 rpb column for this head
  int b = blockIdx.x >> 3, hh = blockIdx.x & 7;
  int tid = threadIdx.x, lane = tid & 31, wave = tid >> 5, hi8 = ((lane >> 4) << 3);
  const size_t base = (size_t)b * BASE_STRIDE;

  for (int idx = tid; idx < 729; idx += 256) rc[idx] = rpbt[idx * NHEAD + hh];
  for (int idx = tid; idx < 208 * 32; idx += 256) {
    int i = idx >> 5, dd = idx & 31;
    Qs[idx] = (_Float16)(i < NN ? qraw[base + (size_t)i * CC + hh * HD + dd] * SCALE : 0.f);
  }
  for (int idx = tid; idx < 32 * 208; idx += 256) {
    int dd = idx / 208, i = idx % 208;
    Kt[idx] = (_Float16)(i < NN ? kraw[base + (size_t)i * CC + hh * HD + dd] : 0.f);
  }
  for (int idx = tid; idx < 224 * 32; idx += 256) {
    int i = idx >> 5, dd = idx & 31;
    Vs[idx] = (_Float16)(i < NN ? vraw[base + (size_t)i * CC + hh * HD + dd] : 0.f);
  }
  __syncthreads();
  // scores: 13x13 tiles, single K-chunk (hd=32)
  for (int t = wave; t < 169; t += 8) {
    int mt = t % 13, nt = t / 13;
    v16h a = ldA(Qs, 32, mt * 16, 0);
    v16h bf = ldB(Kt, 208, 0, nt * 16);
    v8f c = {};
    c = wmma_f32(a, bf, c);
    int j = nt * 16 + (lane & 15);
#pragma unroll
    for (int r = 0; r < 8; ++r) {
      int i = mt * 16 + r + hi8;
      float v = c[r];
      if (i < NN && j < NN) {
        int dy = i / 14 - j / 14 + 13;
        int dx = i % 14 - j % 14 + 13;
        v += rc[dy * 27 + dx];
      }
      Sf[i * 208 + j] = v;
    }
  }
  __syncthreads();
  if (tid < 208) {
    int row = tid;
    _Float16* pr = Pf + row * 224;
    if (row < NN) {
      const float* sr = Sf + row * 208;
      float m = -1e30f;
      for (int j = 0; j < NN; ++j) m = fmaxf(m, sr[j]);
      float s = 0.f;
      for (int j = 0; j < NN; ++j) s += __expf(sr[j] - m);
      float inv = 1.f / s;
      for (int j = 0; j < 224; ++j)
        pr[j] = (_Float16)(j < NN ? __expf(sr[j] - m) * inv : 0.f);
    } else {
      for (int j = 0; j < 224; ++j) pr[j] = (_Float16)0.f;
    }
  }
  __syncthreads();
  // P @ V : 13 m-tiles x 2 n-tiles x 7 K-chunks
  for (int t = wave; t < 26; t += 8) {
    int mt = t % 13, nt = t / 13;
    v8f acc = {};
#pragma unroll
    for (int kc = 0; kc < 7; ++kc) {
      v16h a = ldA(Pf, 224, mt * 16, kc * 32);
      v16h bf = ldB(Vs, 32, kc * 32, nt * 16);
      acc = wmma_f32(a, bf, acc);
    }
    int col = nt * 16 + (lane & 15);
#pragma unroll
    for (int r = 0; r < 8; ++r) {
      int i = mt * 16 + r + hi8;
      if (i < NN) x1[base + (size_t)i * CC + hh * HD + col] = acc[r];
    }
  }
}

// ---------------------------------------------------------------------------
// K7: chan_inter -> sigq[b][c] = sigmoid(cmap)
// ---------------------------------------------------------------------------
__global__ void k_chint(const float* __restrict__ x1, const float* ci1w,
                        const float* ci1b, const float* bng, const float* bnb,
                        const float* ci2w, const float* ci2b, float* __restrict__ sigq) {
  __shared__ float p[CC], tv[RR];
  int b = blockIdx.x, c = threadIdx.x;
  float s = 0.f;
  const float* xp = x1 + (size_t)b * BASE_STRIDE + c;
  for (int i = 0; i < NN; ++i) s += xp[(size_t)i * CC];
  p[c] = s * (1.f / 196.f);
  __syncthreads();
  if (c < RR) {
    float a = ci1b[c];
    for (int j = 0; j < CC; ++j) a += ci1w[(c * CC + j) * 25 + 12] * p[j];
    a = bng[c] * a * 0.99999500003749968f + bnb[c];   // 1/sqrt(1+1e-5)
    tv[c] = fmaxf(a, 0.f);
  }
  __syncthreads();
  float o = ci2b[c];
  for (int r = 0; r < RR; ++r) o += ci2w[(c * RR + r) * 25 + 12] * tv[r];
  sigq[b * CC + c] = sigmoidf(o);
}

// ---------------------------------------------------------------------------
// K8: precompute pb[h][a][n] and ab[h][n][a] (bilinear 7->14 + axial biases)
// ---------------------------------------------------------------------------
__global__ void k_bias(const float* __restrict__ an, const float* __restrict__ na,
                       const float* __restrict__ ah, const float* __restrict__ aw,
                       const float* __restrict__ ha, const float* __restrict__ wa,
                       float* __restrict__ pb, float* __restrict__ ab) {
  int tid = blockIdx.x * 256 + threadIdx.x;
  const int tot = NHEAD * AG * NN;
  if (tid >= 2 * tot) return;
  int which = tid / tot;
  int rem = tid % tot;
  int hh = rem / (AG * NN);
  int a = (rem / NN) % AG;
  int n = rem % NN;
  int y = n / 14, x = n % 14;
  int iy0, iy1, ix0, ix1;
  float ly, lx;
  bgrid(y, iy0, iy1, ly);
  bgrid(x, ix0, ix1, lx);
  const float* src = (which == 0 ? an : na) + ((size_t)hh * AG + a) * 49;
  float v00 = src[iy0 * 7 + ix0], v10 = src[iy1 * 7 + ix0];
  float v01 = src[iy0 * 7 + ix1], v11 = src[iy1 * 7 + ix1];
  float bv = (v00 * (1.f - ly) + v10 * ly) * (1.f - lx) +
             (v01 * (1.f - ly) + v11 * ly) * lx;
  if (which == 0) {
    bv += ah[((size_t)hh * AG + a) * 14 + y] + aw[((size_t)hh * AG + a) * 14 + x];
    pb[((size_t)hh * AG + a) * NN + n] = bv;
  } else {
    bv += ha[((size_t)hh * 14 + y) * AG + a] + wa[((size_t)hh * 14 + x) * AG + a];
    ab[((size_t)hh * NN + n) * AG + a] = bv;
  }
}

// ---------------------------------------------------------------------------
// K9: agent attention. One WG per (b, head). All 6 GEMMs through WMMA.
// ---------------------------------------------------------------------------
__global__ void __launch_bounds__(256) k_agent(const float* __restrict__ qraw,
                                               const float* __restrict__ kraw,
                                               const float* __restrict__ vraw,
                                               const float* __restrict__ vc,
                                               const float* __restrict__ at,
                                               const float* __restrict__ atk,
                                               const float* __restrict__ pb,
                                               const float* __restrict__ ab,
                                               float* __restrict__ xo) {
  extern __shared__ char smem[];
  _Float16* Ats   = (_Float16*)smem;     // 64x32  (at * scale, padded)
  _Float16* At_t  = Ats + 64 * 32;       // 32x64  transposed
  _Float16* Atks  = At_t + 32 * 64;      // 64x32
  _Float16* Atk_t = Atks + 64 * 32;      // 32x64
  _Float16* Qs    = Atk_t + 32 * 64;     // 208x32 (q_raw*scale)
  _Float16* Qt    = Qs + 208 * 32;       // 32x208
  _Float16* Ks    = Qt + 32 * 208;       // 208x32
  _Float16* Kt    = Ks + 208 * 32;       // 32x208
  _Float16* Vs    = Kt + 32 * 208;       // 224x32
  _Float16* V1s   = Vs + 224 * 32;       // 224x32 (from vc)
  _Float16* Avs   = V1s + 224 * 32;      // 64x32 agent_v (f16)
  _Float16* P1    = Avs + 64 * 32;       // 64x224 probs (reused later as P3 208x64)
  _Float16* P2    = P1 + 64 * 224;       // 64x224
  float* Sb       = (float*)(P2 + 64 * 224);  // 13312 f32 logits (reused)
  int b = blockIdx.x >> 3, hh = blockIdx.x & 7;
  int tid = threadIdx.x, lane = tid & 31, wave = tid >> 5, hi8 = ((lane >> 4) << 3);
  const size_t base = (size_t)b * BASE_STRIDE;

  for (int idx = tid; idx < 64 * 32; idx += 256) {
    int a_ = idx >> 5, dd = idx & 31;
    float va = a_ < AG ? at[((size_t)b * AG + a_) * CC + hh * HD + dd] * SCALE : 0.f;
    float vk = a_ < AG ? atk[((size_t)b * AG + a_) * CC + hh * HD + dd] * SCALE : 0.f;
    Ats[idx] = (_Float16)va;  At_t[dd * 64 + a_] = (_Float16)va;
    Atks[idx] = (_Float16)vk; Atk_t[dd * 64 + a_] = (_Float16)vk;
  }
  for (int idx = tid; idx < 208 * 32; idx += 256) {
    int i = idx >> 5, dd = idx & 31;
    float vq = i < NN ? qraw[base + (size_t)i * CC + hh * HD + dd] * SCALE : 0.f;
    float vk = i < NN ? kraw[base + (size_t)i * CC + hh * HD + dd] : 0.f;
    Qs[idx] = (_Float16)vq; Qt[dd * 208 + i] = (_Float16)vq;
    Ks[idx] = (_Float16)vk; Kt[dd * 208 + i] = (_Float16)vk;
  }
  for (int idx = tid; idx < 224 * 32; idx += 256) {
    int i = idx >> 5, dd = idx & 31;
    Vs[idx]  = (_Float16)(i < NN ? vraw[base + (size_t)i * CC + hh * HD + dd] : 0.f);
    V1s[idx] = (_Float16)(i < NN ? vc[((size_t)b * CC + hh * HD + dd) * NN + i] : 0.f);
  }
  __syncthreads();
  // --- S1 = (at*scale) @ K^T + pb
  for (int t = wave; t < 52; t += 8) {
    int mt = t / 13, nt = t % 13;
    v8f c = {};
    c = wmma_f32(ldA(Ats, 32, mt * 16, 0), ldB(Kt, 208, 0, nt * 16), c);
    int j = nt * 16 + (lane & 15);
#pragma unroll
    for (int r = 0; r < 8; ++r) {
      int ar = mt * 16 + r + hi8;
      float v = c[r];
      if (ar < AG && j < NN) v += pb[((size_t)hh * AG + ar) * NN + j];
      Sb[ar * 208 + j] = v;
    }
  }
  __syncthreads();
  if (tid < 64) {
    int row = tid;
    _Float16* pr = P1 + row * 224;
    if (row < AG) {
      const float* sr = Sb + row * 208;
      float m = -1e30f;
      for (int j = 0; j < NN; ++j) m = fmaxf(m, sr[j]);
      float s = 0.f;
      for (int j = 0; j < NN; ++j) s += __expf(sr[j] - m);
      float inv = 1.f / s;
      for (int j = 0; j < 224; ++j) pr[j] = (_Float16)(j < NN ? __expf(sr[j] - m) * inv : 0.f);
    } else for (int j = 0; j < 224; ++j) pr[j] = (_Float16)0.f;
  }
  __syncthreads();
  // --- S2 = (atk*scale) @ Q^T + pb
  for (int t = wave; t < 52; t += 8) {
    int mt = t / 13, nt = t % 13;
    v8f c = {};
    c = wmma_f32(ldA(Atks, 32, mt * 16, 0), ldB(Qt, 208, 0, nt * 16), c);
    int j = nt * 16 + (lane & 15);
#pragma unroll
    for (int r = 0; r < 8; ++r) {
      int ar = mt * 16 + r + hi8;
      float v = c[r];
      if (ar < AG && j < NN) v += pb[((size_t)hh * AG + ar) * NN + j];
      Sb[ar * 208 + j] = v;
    }
  }
  __syncthreads();
  if (tid < 64) {
    int row = tid;
    _Float16* pr = P2 + row * 224;
    if (row < AG) {
      const float* sr = Sb + row * 208;
      float m = -1e30f;
      for (int j = 0; j < NN; ++j) m = fmaxf(m, sr[j]);
      float s = 0.f;
      for (int j = 0; j < NN; ++j) s += __expf(sr[j] - m);
      float inv = 1.f / s;
      for (int j = 0; j < 224; ++j) pr[j] = (_Float16)(j < NN ? __expf(sr[j] - m) * inv : 0.f);
    } else for (int j = 0; j < 224; ++j) pr[j] = (_Float16)0.f;
  }
  __syncthreads();
  // --- agent_v = P1 @ V1 + P2 @ V   (4 m-tiles x 2 n-tiles)
  for (int t = wave; t < 8; t += 8) {
    int mt = t >> 1, nt = t & 1;
    v8f acc = {};
#pragma unroll
    for (int kc = 0; kc < 7; ++kc) {
      acc = wmma_f32(ldA(P1, 224, mt * 16, kc * 32), ldB(V1s, 32, kc * 32, nt * 16), acc);
      acc = wmma_f32(ldA(P2, 224, mt * 16, kc * 32), ldB(Vs, 32, kc * 32, nt * 16), acc);
    }
    int col = nt * 16 + (lane & 15);
#pragma unroll
    for (int r = 0; r < 8; ++r) Avs[(mt * 16 + r + hi8) * 32 + col] = (_Float16)acc[r];
  }
  __syncthreads();
  // --- S3 = Q @ (at*scale)^T + ab  (q_attn logits)
  for (int t = wave; t < 52; t += 8) {
    int mt = t % 13, nt = t / 13;
    v8f c = {};
    c = wmma_f32(ldA(Qs, 32, mt * 16, 0), ldB(At_t, 64, 0, nt * 16), c);
    int acol = nt * 16 + (lane & 15);
#pragma unroll
    for (int r = 0; r < 8; ++r) {
      int i = mt * 16 + r + hi8;
      float v = c[r];
      if (i < NN && acol < AG) v += ab[((size_t)hh * NN + i) * AG + acol];
      Sb[i * 64 + acol] = v;
    }
  }
  __syncthreads();
  if (tid < 208) {          // softmax over 49 agents -> P3 (reuse P1, stride 64)
    int row = tid;
    _Float16* pr = P1 + row * 64;
    if (row < NN) {
      const float* sr = Sb + row * 64;
      float m = -1e30f;
      for (int j = 0; j < AG; ++j) m = fmaxf(m, sr[j]);
      float s = 0.f;
      for (int j = 0; j < AG; ++j) s += __expf(sr[j] - m);
      float inv = 1.f / s;
      for (int j = 0; j < 64; ++j) pr[j] = (_Float16)(j < AG ? __expf(sr[j] - m) * inv : 0.f);
    } else for (int j = 0; j < 64; ++j) pr[j] = (_Float16)0.f;
  }
  __syncthreads();
  // --- S4 = K @ (atk*scale)^T + ab  (k_attn logits)
  for (int t = wave; t < 52; t += 8) {
    int mt = t % 13, nt = t / 13;
    v8f c = {};
    c = wmma_f32(ldA(Ks, 32, mt * 16, 0), ldB(Atk_t, 64, 0, nt * 16), c);
    int acol = nt * 16 + (lane & 15);
#pragma unroll
    for (int r = 0; r < 8; ++r) {
      int i = mt * 16 + r + hi8;
      float v = c[r];
      if (i < NN && acol < AG) v += ab[((size_t)hh * NN + i) * AG + acol];
      Sb[i * 64 + acol] = v;
    }
  }
  __syncthreads();
  if (tid < 208) {          // softmax, ACCUMULATE into P3
    int row = tid;
    if (row < NN) {
      const float* sr = Sb + row * 64;
      _Float16* pr = P1 + row * 64;
      float m = -1e30f;
      for (int j = 0; j < AG; ++j) m = fmaxf(m, sr[j]);
      float s = 0.f;
      for (int j = 0; j < AG; ++j) s += __expf(sr[j] - m);
      float inv = 1.f / s;
      for (int j = 0; j < AG; ++j)
        pr[j] = (_Float16)((float)pr[j] + __expf(sr[j] - m) * inv);
    }
  }
  __syncthreads();
  // --- xo = (q_attn + k_attn) @ agent_v : 13 m-tiles x 2 n-tiles x 2 K-chunks
  for (int t = wave; t < 26; t += 8) {
    int mt = t % 13, nt = t / 13;
    v8f acc = {};
#pragma unroll
    for (int kc = 0; kc < 2; ++kc)
      acc = wmma_f32(ldA(P1, 64, mt * 16, kc * 32), ldB(Avs, 32, kc * 32, nt * 16), acc);
    int col = nt * 16 + (lane & 15);
#pragma unroll
    for (int r = 0; r < 8; ++r) {
      int i = mt * 16 + r + hi8;
      if (i < NN) xo[base + (size_t)i * CC + hh * HD + col] = acc[r];
    }
  }
}

// ---------------------------------------------------------------------------
// K10: two depthwise 5x5 convs (conv_x2 path + qc path), accumulated into xo
// ---------------------------------------------------------------------------
__global__ void k_dwc(const float* __restrict__ qraw, const float* __restrict__ sigq,
                      const float* __restrict__ dw, const float* __restrict__ db,
                      float* __restrict__ xo) {
  __shared__ float w5[25];
  __shared__ float p1[NN], p2[NN];
  int b = blockIdx.x / CC, cc = blockIdx.x % CC;
  int t = threadIdx.x;
  if (t < 25) w5[t] = dw[cc * 25 + t];
  float sg = sigq[b * CC + cc];
  const size_t base = (size_t)b * BASE_STRIDE;
  if (t < NN) {
    p2[t] = SCALE * qraw[base + (size_t)t * CC + cc];     // qc plane
    int idx = t * CC + cc;                                // conv_x2 plane (scrambled q)
    int hh = idx / 6272;
    int rem = idx % 6272;
    int i = rem >> 5, dd = rem & 31;
    p1[t] = sg * SCALE * qraw[base + (size_t)i * CC + hh * HD + dd];
  }
  __syncthreads();
  if (t < NN) {
    int y = t / 14, x = t % 14;
    float a1 = 0.f, a2 = 0.f;
#pragma unroll
    for (int ky = 0; ky < 5; ++ky) {
      int iy = y + ky - 2;
      if (iy < 0 || iy >= 14) continue;
#pragma unroll
      for (int kx = 0; kx < 5; ++kx) {
        int ix = x + kx - 2;
        if (ix < 0 || ix >= 14) continue;
        float wv = w5[ky * 5 + kx];
        a1 += p1[iy * 14 + ix] * wv;
        a2 += p2[iy * 14 + ix] * wv;
      }
    }
    xo[base + (size_t)t * CC + cc] += a1 + a2 + 2.f * db[cc];
  }
}

// ---------------------------------------------------------------------------
// K11: final projection out = xo @ W^T + b  (WMMA GEMM, one WG per batch)
// ---------------------------------------------------------------------------
__global__ void __launch_bounds__(256) k_proj(const float* __restrict__ xo,
                                              const float* __restrict__ pw,
                                              const float* __restrict__ pbias,
                                              float* __restrict__ out) {
  extern __shared__ char smem[];
  _Float16* Xs = (_Float16*)smem;      // 208x256
  _Float16* Wp = Xs + 208 * 256;       // 256x256, K-major: Wp[j][o] = W[o][j]
  int b = blockIdx.x;
  int tid = threadIdx.x, lane = tid & 31, wave = tid >> 5, hi8 = ((lane >> 4) << 3);
  const size_t base = (size_t)b * BASE_STRIDE;
  for (int idx = tid; idx < 208 * 256; idx += 256) {
    int i = idx >> 8, j = idx & 255;
    Xs[idx] = (_Float16)(i < NN ? xo[base + (size_t)i * CC + j] : 0.f);
  }
  for (int idx = tid; idx < 256 * 256; idx += 256) {
    int j = idx >> 8, o = idx & 255;
    __builtin_prefetch(&pw[(size_t)o * CC + j], 0, 1);
    Wp[idx] = (_Float16)pw[(size_t)o * CC + j];
  }
  __syncthreads();
  for (int t = wave; t < 208; t += 8) {
    int mt = t % 13, nt = t / 13;
    v8f acc = {};
#pragma unroll
    for (int kc = 0; kc < 8; ++kc)
      acc = wmma_f32(ldA(Xs, 256, mt * 16, kc * 32), ldB(Wp, 256, kc * 32, nt * 16), acc);
    int o = nt * 16 + (lane & 15);
    float bias = pbias[o];
#pragma unroll
    for (int r = 0; r < 8; ++r) {
      int i = mt * 16 + r + hi8;
      if (i < NN) out[base + (size_t)i * CC + o] = acc[r] + bias;
    }
  }
}

// ---------------------------------------------------------------------------
// Launcher
// ---------------------------------------------------------------------------
extern "C" void kernel_launch(void* const* d_in, const int* in_sizes, int n_in,
                              void* d_out, int out_size, void* d_ws, size_t ws_size,
                              hipStream_t stream) {
  (void)in_sizes; (void)n_in; (void)out_size; (void)ws_size;
  const float* x       = (const float*)d_in[0];
  const float* lka_w   = (const float*)d_in[3];
  const float* lka_b   = (const float*)d_in[4];
  const float* conv5_w = (const float*)d_in[5];
  const float* conv5_b = (const float*)d_in[6];
  const float* caa1_w  = (const float*)d_in[7];
  const float* caa1_b  = (const float*)d_in[8];
  const float* caa2_w  = (const float*)d_in[9];
  const float* caa2_b  = (const float*)d_in[10];
  const float* cam1_w  = (const float*)d_in[11];
  const float* cam1_b  = (const float*)d_in[12];
  const float* cam2_w  = (const float*)d_in[13];
  const float* cam2_b  = (const float*)d_in[14];
  const float* sa_w    = (const float*)d_in[15];
  const float* sa_b    = (const float*)d_in[16];
  const float* ci1_w   = (const float*)d_in[17];
  const float* ci1_b   = (const float*)d_in[18];
  const float* bn_g    = (const float*)d_in[19];
  const float* bn_b    = (const float*)d_in[20];
  const float* ci2_w   = (const float*)d_in[21];
  const float* ci2_b   = (const float*)d_in[22];
  const float* rpb_t   = (const float*)d_in[23];
  const float* an_bias = (const float*)d_in[24];
  const float* na_bias = (const float*)d_in[25];
  const float* ah_bias = (const float*)d_in[26];
  const float* aw_bias = (const float*)d_in[27];
  const float* ha_bias = (const float*)d_in[28];
  const float* wa_bias = (const float*)d_in[29];
  const float* dwc_w   = (const float*)d_in[30];
  const float* dwc_b   = (const float*)d_in[31];
  const float* proj_w  = (const float*)d_in[32];
  const float* proj_b  = (const float*)d_in[33];
  float* out = (float*)d_out;

  // workspace carve (floats)
  float* ws    = (float*)d_ws;
  const size_t PLANE = (size_t)BB * CC * NN;   // 6,422,528
  float* q_raw = ws;
  float* k_raw = q_raw + PLANE;
  float* v_raw = k_raw + PLANE;
  float* v5    = v_raw + PLANE;                // becomes vc after CBAM
  float* x1    = v5 + PLANE;
  float* xo    = x1 + PLANE;
  float* at    = xo + PLANE;
  float* atk   = at + (size_t)BB * AG * CC;
  float* sigc  = atk + (size_t)BB * AG * CC;
  float* sigq  = sigc + (size_t)BB * CC;
  float* pbuf  = sigq + (size_t)BB * CC;
  float* abuf  = pbuf + (size_t)NHEAD * AG * NN;

  const int SMEM_CONV5 = (324 * 256 + 256 * 32) * 2;                       // 182,272
  const int SMEM_ATTN  = (208 * 32 + 32 * 208 + 224 * 32 + 208 * 224) * 2
                         + (208 * 208 + 729) * 4;                          // 310,116
  const int SMEM_AGENT = (2 * (64 * 32) + 2 * (32 * 64) + 2 * (208 * 32)
                          + 2 * (32 * 208) + 2 * (224 * 32) + 64 * 32
                          + 2 * (64 * 224)) * 2 + 13312 * 4;               // 212,992
  const int SMEM_PROJ  = (208 * 256 + 256 * 256) * 2;                      // 237,568

  (void)hipFuncSetAttribute((const void*)k_conv5,
      hipFuncAttributeMaxDynamicSharedMemorySize, SMEM_CONV5);
  (void)hipFuncSetAttribute((const void*)k_attn,
      hipFuncAttributeMaxDynamicSharedMemorySize, SMEM_ATTN);
  (void)hipFuncSetAttribute((const void*)k_agent,
      hipFuncAttributeMaxDynamicSharedMemorySize, SMEM_AGENT);
  (void)hipFuncSetAttribute((const void*)k_proj,
      hipFuncAttributeMaxDynamicSharedMemorySize, SMEM_PROJ);

  k_bias<<<(2 * NHEAD * AG * NN + 255) / 256, 256, 0, stream>>>(
      an_bias, na_bias, ah_bias, aw_bias, ha_bias, wa_bias, pbuf, abuf);
  k_lka<<<BB * 3 * CC, 256, 0, stream>>>(x, lka_w, lka_b, q_raw, k_raw, v_raw);
  k_pool<<<(BB * AG * CC + 255) / 256, 256, 0, stream>>>(q_raw, k_raw, at, atk);
  k_conv5<<<BB * 8, 256, SMEM_CONV5, stream>>>(v_raw, conv5_w, conv5_b, v5);
  k_cbam_ch<<<BB, 256, 0, stream>>>(v5, caa1_w, caa1_b, caa2_w, caa2_b,
                                    cam1_w, cam1_b, cam2_w, cam2_b, sigc);
  k_cbam_sp<<<BB, 256, 0, stream>>>(v5, sigc, sa_w, sa_b);
  k_attn<<<BB * NHEAD, 256, SMEM_ATTN, stream>>>(q_raw, k_raw, v_raw, rpb_t, x1);
  k_chint<<<BB, 256, 0, stream>>>(x1, ci1_w, ci1_b, bn_g, bn_b, ci2_w, ci2_b, sigq);
  k_agent<<<BB * NHEAD, 256, SMEM_AGENT, stream>>>(q_raw, k_raw, v_raw, v5,
                                                   at, atk, pbuf, abuf, xo);
  k_dwc<<<BB * CC, 256, 0, stream>>>(q_raw, sigq, dwc_w, dwc_b, xo);
  k_proj<<<BB, 256, SMEM_PROJ, stream>>>(xo, proj_w, proj_b, out);
}
